// TimeSeriesTransformer_24378234372649
// MI455X (gfx1250) — matrix operand order repaired and checked by reading, not compiled, not run on hardware
//
#include <hip/hip_runtime.h>
#include <hip/hip_bf16.h>
#include <cmath>

// ---------------------------------------------------------------------------
// Non-stationary Transformer forward for gfx1250 (MI455X).
// Dense GEMMs + attention matmuls: v_wmma_f32_16x16x32_f16.
// A/K tile staging: global_load_async_to_lds_b128 (ASYNCcnt) + double buffer.
//
// d_in flattening assumption (setup_inputs() dict insertion order, recursive):
//  0: x_enc  1: y_dec
//  2: W_enc_in [7,512]  3: b_enc_in  4: W_dec_in  5: b_dec_in
//  6..21 : enc[0] {Wq,bq,Wk,bk,Wv,bv,Wo,bo,Wf1,bf1,Wf2,bf2,ln1_g,ln1_b,ln2_g,ln2_b}
// 22..37 : enc[1] (same)
// 38: enc_norm_g  39: enc_norm_b
// 40..47: dec.sa {Wq,bq,Wk,bk,Wv,bv,Wo,bo}   48..55: dec.ca (same)
// 56: dec.Wf1 57: dec.bf1 58: dec.Wf2 59: dec.bf2
// 60..65: ln1_g,ln1_b,ln2_g,ln2_b,ln3_g,ln3_b
// 66: dec_norm_g 67: dec_norm_b 68: Wp 69: bp
// 70..75: tau proj   {conv_w, W1, b1, W2, b2, W3[256,1]}
// 76..81: delta proj {conv_w, W1, b1, W2, b2, W3[256,512]}
// ---------------------------------------------------------------------------

typedef __attribute__((ext_vector_type(16))) _Float16 v16h;
typedef __attribute__((ext_vector_type(8)))  float    v8f;

union AF { v16h v; unsigned u[8]; };
union CF { v8f  v; float    f[8]; };
union U4 { uint4 q; unsigned short s[8]; };

#define WMMA_F16(a, b, c) \
  __builtin_amdgcn_wmma_f32_16x16x32_f16(false, (a), false, (b), (short)0, (c), false, false)

// async 16B copy global -> LDS (per lane), tracked by ASYNCcnt
__device__ __forceinline__ void async_b128(const _Float16* lds_ptr, const _Float16* gptr) {
  unsigned loff = (unsigned)(size_t)lds_ptr;
  unsigned long long ga = (unsigned long long)(size_t)gptr;
  asm volatile("global_load_async_to_lds_b128 %0, %1, off"
               :: "v"(loff), "v"(ga) : "memory");
}
__device__ __forceinline__ void wait_async0() {
  asm volatile("s_wait_asynccnt 0x0" ::: "memory");
}

// ------------------------- f32 -> f16 convert ------------------------------
__global__ void k_f2h(const float* __restrict__ src, _Float16* __restrict__ dst, int n) {
  for (int i = blockIdx.x * blockDim.x + threadIdx.x; i < n; i += gridDim.x * blockDim.x)
    dst[i] = (_Float16)src[i];
}

// ------------------------- preprocessing -----------------------------------
__global__ void k_prep(const float* __restrict__ x_enc, const float* __restrict__ y_dec,
                       float* __restrict__ xdash, float* __restrict__ xdashdec,
                       float* __restrict__ xlag, float* __restrict__ meanb,
                       float* __restrict__ stdb) {
  int id = blockIdx.x * blockDim.x + threadIdx.x;
  if (id >= 32 * 7) return;
  int b = id / 7, i = id % 7;
  const float* x = x_enc + (size_t)b * 512 * 7 + i;
  float s = 0.f, s2 = 0.f, prev = x[0];
  for (int t = 0; t < 512; ++t) {
    float cur = x[(size_t)t * 7];
    xlag [((size_t)b * 512 + t) * 7 + i] = (t == 0) ? cur : prev;
    xdash[((size_t)b * 512 + t) * 7 + i] = (t == 0) ? 0.f : (cur - prev);
    s += cur; s2 += cur * cur;
    prev = cur;
  }
  float mu  = s / 512.f;
  float var = s2 / 512.f - mu * mu;
  meanb[b * 7 + i] = mu;
  stdb [b * 7 + i] = sqrtf(var + 1e-5f);
  for (int t = 0; t < 256; ++t)
    xdashdec[((size_t)b * 768 + t) * 7 + i] = xdash[((size_t)b * 512 + 256 + t) * 7 + i];
  for (int t = 256; t < 768; ++t)
    xdashdec[((size_t)b * 768 + t) * 7 + i] = y_dec[((size_t)b * 512 + (t - 256)) * 7 + i];
}

// ------------------------- tau / delta projector ---------------------------
__global__ __launch_bounds__(256)
void k_proj(const float* __restrict__ x_raw, const float* __restrict__ stats,
            const float* __restrict__ conv_w,
            const float* __restrict__ W1, const float* __restrict__ b1,
            const float* __restrict__ W2, const float* __restrict__ b2,
            const float* __restrict__ W3, float* __restrict__ out,
            int out_dim, int do_exp) {
  __shared__ float convs[7];
  __shared__ float h[14];
  __shared__ float h1[256];
  __shared__ float h2[256];
  int b = blockIdx.x, t = threadIdx.x;
  if (t < 7) convs[t] = 0.f;
  __syncthreads();
  float acc[7] = {0.f, 0.f, 0.f, 0.f, 0.f, 0.f, 0.f};
  for (int c = t; c < 512; c += 256) {
    float w0 = conv_w[c * 3 + 0], w1 = conv_w[c * 3 + 1], w2 = conv_w[c * 3 + 2];
    const float* xr = x_raw + ((size_t)b * 512 + c) * 7;
    float xp[9];
    xp[0] = xr[6];
    for (int j = 0; j < 7; ++j) xp[1 + j] = xr[j];
    xp[8] = xr[0];
    for (int e = 0; e < 7; ++e) acc[e] += xp[e] * w0 + xp[e + 1] * w1 + xp[e + 2] * w2;
  }
  for (int e = 0; e < 7; ++e) atomicAdd(&convs[e], acc[e]);
  __syncthreads();
  if (t < 7) { h[t] = convs[t]; h[7 + t] = stats[b * 7 + t]; }
  __syncthreads();
  float a1 = b1[t];
  for (int j = 0; j < 14; ++j) a1 += h[j] * W1[j * 256 + t];
  h1[t] = fmaxf(a1, 0.f);
  __syncthreads();
  float a2 = b2[t];
  for (int j = 0; j < 256; ++j) a2 += h1[j] * W2[j * 256 + t];
  h2[t] = fmaxf(a2, 0.f);
  __syncthreads();
  for (int o = t; o < out_dim; o += 256) {
    float a3 = 0.f;
    for (int j = 0; j < 256; ++j) a3 += h2[j] * W3[j * out_dim + o];
    out[(size_t)b * out_dim + o] = do_exp ? __expf(a3) : a3;
  }
}

// ------------------------- input embedding + PE ----------------------------
__global__ void k_embed(const float* __restrict__ xin, const float* __restrict__ W,
                        const float* __restrict__ bias, float* __restrict__ outf,
                        _Float16* __restrict__ outh, int Mrows, int L) {
  int total = Mrows * 512;
  for (int idx = blockIdx.x * blockDim.x + threadIdx.x; idx < total;
       idx += gridDim.x * blockDim.x) {
    int m = idx >> 9, d = idx & 511;
    int l = m % L;
    float acc = bias[d];
    const float* xr = xin + (size_t)m * 7;
    #pragma unroll
    for (int i = 0; i < 7; ++i) acc += xr[i] * W[i * 512 + d];
    int j = d >> 1;
    float freq = __expf(-(float)(2 * j) * (9.210340371976184f / 512.0f));
    float ang = (float)l * freq;
    acc += (d & 1) ? cosf(ang) : sinf(ang);
    outf[idx] = acc;
    outh[idx] = (_Float16)acc;
  }
}

// ------------------------- WMMA GEMM ---------------------------------------
// C[M,N] = A[M,K](f16) @ W[K,N](f16) + bias (+resid) [relu] -> f32/f16.
// 4 waves / block, 128x64 C tile, each wave 32x64 (8 WMMA per K-step of 32).
// A tile: async-to-LDS double-buffered; B tile: register-relay transpose,
// double-buffered. One barrier per K-step.
template <int RESID, int RELU, int OUTF, int OUTH>
__global__ __launch_bounds__(128)
void k_gemm(const _Float16* __restrict__ A, const _Float16* __restrict__ W,
            const float* __restrict__ bias, const float* __restrict__ resid,
            float* __restrict__ outf, _Float16* __restrict__ outh,
            int M, int N, int K) {
  __shared__ _Float16 sA[2][128 * 40];  // [m][k], 80B rows (16B-aligned)
  __shared__ _Float16 sB[2][64 * 40];   // transposed [n][k]
  const int tid = threadIdx.x;
  const int lane = tid & 31, wv = tid >> 5;
  const int half = lane >> 4, l16 = lane & 15;
  const int m0 = blockIdx.y * 128, n0 = blockIdx.x * 64;
  (void)M;

  const int arow = tid >> 2, acol = tid & 3;   // A: 4 async b128 / thread / step
  const int bkr = tid >> 3, bnc = tid & 7;     // B: 2 b128 / thread / step
  const _Float16* ap = A + (size_t)(m0 + arow) * K + 8 * acol;
  const _Float16* wp = W + (size_t)bkr * N + n0 + 8 * bnc;

  CF acc[2][4];
  #pragma unroll
  for (int ms = 0; ms < 2; ++ms)
    #pragma unroll
    for (int nf = 0; nf < 4; ++nf)
      #pragma unroll
      for (int i = 0; i < 8; ++i) acc[ms][nf].f[i] = 0.f;

  // prologue: stage tile 0 into buffer 0
  #pragma unroll
  for (int it = 0; it < 4; ++it)
    async_b128(&sA[0][(arow + it * 32) * 40 + 8 * acol], ap + (size_t)it * 32 * K);
  {
    U4 bv[2];
    #pragma unroll
    for (int it = 0; it < 2; ++it)
      bv[it].q = *(const uint4*)(wp + (size_t)(it * 16) * N);
    #pragma unroll
    for (int it = 0; it < 2; ++it) {
      int kr = bkr + it * 16;
      #pragma unroll
      for (int j = 0; j < 8; ++j)
        *(unsigned short*)(&sB[0][(8 * bnc + j) * 40 + kr]) = bv[it].s[j];
    }
  }

  for (int k0 = 0; k0 < K; k0 += 32) {
    const int cur = (k0 >> 5) & 1, nxt = cur ^ 1;
    const int kn = k0 + 32;
    U4 bv[2];
    if (kn < K) {
      #pragma unroll
      for (int it = 0; it < 2; ++it)
        bv[it].q = *(const uint4*)(wp + (size_t)(kn + it * 16) * N);
    }
    wait_async0();        // this wave's async writes of tile k0 complete
    __syncthreads();      // everyone done reading buffer nxt; tile k0 visible
    if (kn < K) {
      #pragma unroll
      for (int it = 0; it < 4; ++it)
        async_b128(&sA[nxt][(arow + it * 32) * 40 + 8 * acol],
                   ap + (size_t)it * 32 * K + kn);
      #pragma unroll
      for (int it = 0; it < 2; ++it) {
        int kr = bkr + it * 16;
        #pragma unroll
        for (int j = 0; j < 8; ++j)
          *(unsigned short*)(&sB[nxt][(8 * bnc + j) * 40 + kr]) = bv[it].s[j];
      }
    }

    AF a[2];
    #pragma unroll
    for (int ms = 0; ms < 2; ++ms) {
      int am = wv * 32 + ms * 16 + l16;
      #pragma unroll
      for (int i = 0; i < 8; ++i) {
        int k = 2 * i + 8 * half + (i >= 4 ? 8 : 0);
        a[ms].u[i] = *(const unsigned*)(&sA[cur][am * 40 + k]);
      }
    }
    #pragma unroll
    for (int nf = 0; nf < 4; ++nf) {
      AF bf;
      int bn = nf * 16 + l16;
      #pragma unroll
      for (int i = 0; i < 8; ++i) {
        int k = 2 * i + 8 * half + (i >= 4 ? 8 : 0);
        bf.u[i] = *(const unsigned*)(&sB[cur][bn * 40 + k]);
      }
      acc[0][nf].v = WMMA_F16(a[0].v, bf.v, acc[0][nf].v);
      acc[1][nf].v = WMMA_F16(a[1].v, bf.v, acc[1][nf].v);
    }
  }
  // epilogue (straight-line; flags are compile-time)
  #pragma unroll
  for (int ms = 0; ms < 2; ++ms)
    #pragma unroll
    for (int nf = 0; nf < 4; ++nf) {
      int col = n0 + nf * 16 + l16;
      float bc = bias[col];
      #pragma unroll
      for (int i = 0; i < 8; ++i) {
        int row = m0 + wv * 32 + ms * 16 + i + 8 * half;
        float v = acc[ms][nf].f[i] + bc;
        if (RESID) v += resid[(size_t)row * N + col];
        if (RELU) v = fmaxf(v, 0.f);
        if (OUTF) outf[(size_t)row * N + col] = v;
        if (OUTH) outh[(size_t)row * N + col] = (_Float16)v;
      }
    }
}

// ------------------------- fused de-stationary attention -------------------
// softmax((Q K^T * tau + delta) / sqrt(E)) @ V, flash-style, per (b,h,64 rows).
// K tiles: async-to-LDS double-buffered; V tiles: register-relay transpose.
template <int CAUSAL, int HASD>
__global__ __launch_bounds__(128)
void k_attn(const _Float16* __restrict__ Q, const _Float16* __restrict__ Kk,
            const _Float16* __restrict__ Vv, _Float16* __restrict__ Oo,
            const float* __restrict__ tau, const float* __restrict__ delta,
            int H, int Lq, int S) {
  const int D = 512;
  __shared__ _Float16 sK [2][32 * 72];    // [s][e], 144B rows
  __shared__ _Float16 sVt[2][64 * 40];    // transposed [e][s]
  __shared__ _Float16 sP [4][16 * 40];    // per-wave P tile [m][s]
  const int tid = threadIdx.x;
  const int lane = tid & 31, wv = tid >> 5;
  const int half = lane >> 4, l16 = lane & 15;
  const int bh = blockIdx.y, b = bh / H, h = bh % H;
  const int l0 = blockIdx.x * 64;
  const float taub = tau[b];
  const float rsE = 0.125f;  // 1/sqrt(64)

  const int srow = tid >> 2, scol4 = tid & 3;  // one b128 per thread per tile
  const _Float16* kbase = Kk + ((size_t)(b * S + srow)) * D + h * 64 + 8 * scol4;
  const _Float16* vbase = Vv + ((size_t)(b * S + srow)) * D + h * 64 + 8 * scol4;

  // Q fragments (E=64 -> two K=32 chunks), held in registers
  AF qf[2];
  {
    int qrow = l0 + wv * 16 + l16;
    const _Float16* qb = Q + ((size_t)(b * Lq + qrow)) * D + h * 64;
    #pragma unroll
    for (int ch = 0; ch < 2; ++ch)
      #pragma unroll
      for (int i = 0; i < 8; ++i) {
        int k = 2 * i + 8 * half + (i >= 4 ? 8 : 0) + 32 * ch;
        qf[ch].u[i] = *(const unsigned*)(qb + k);
      }
  }

  CF o[4];
  float rmax[8], rsum[8];
  #pragma unroll
  for (int i = 0; i < 8; ++i) { rmax[i] = -INFINITY; rsum[i] = 0.f; }
  #pragma unroll
  for (int nf = 0; nf < 4; ++nf)
    #pragma unroll
    for (int i = 0; i < 8; ++i) o[nf].f[i] = 0.f;

  const int Send = CAUSAL ? (S < l0 + 64 ? S : l0 + 64) : S;

  // prologue: stage tile 0 into buffer 0
  async_b128(&sK[0][srow * 72 + 8 * scol4], kbase);
  {
    U4 vq;
    vq.q = *(const uint4*)vbase;
    #pragma unroll
    for (int j = 0; j < 8; ++j)
      *(unsigned short*)(&sVt[0][(8 * scol4 + j) * 40 + srow]) = vq.s[j];
  }

  for (int s0 = 0; s0 < Send; s0 += 32) {
    const int cur = (s0 >> 5) & 1, nxt = cur ^ 1;
    const int sn = s0 + 32;
    U4 vq;
    if (sn < Send) vq.q = *(const uint4*)(vbase + (size_t)sn * D);
    wait_async0();
    __syncthreads();
    if (sn < Send) {
      async_b128(&sK[nxt][srow * 72 + 8 * scol4], kbase + (size_t)sn * D);
      #pragma unroll
      for (int j = 0; j < 8; ++j)
        *(unsigned short*)(&sVt[nxt][(8 * scol4 + j) * 40 + srow]) = vq.s[j];
    }
    // scores: two 16-col subtiles, each = Q(16x64) @ K^T(64x16)
    CF scf[2];
    #pragma unroll
    for (int t = 0; t < 2; ++t) {
      AF b0, b1;
      int scol = t * 16 + l16;
      #pragma unroll
      for (int i = 0; i < 8; ++i) {
        int e0 = 2 * i + 8 * half + (i >= 4 ? 8 : 0);
        b0.u[i] = *(const unsigned*)(&sK[cur][scol * 72 + e0]);
        b1.u[i] = *(const unsigned*)(&sK[cur][scol * 72 + e0 + 32]);
      }
      CF z;
      #pragma unroll
      for (int i = 0; i < 8; ++i) z.f[i] = 0.f;
      z.v = WMMA_F16(qf[0].v, b0.v, z.v);
      z.v = WMMA_F16(qf[1].v, b1.v, z.v);
      scf[t] = z;
    }
    // de-stationary rescale + online softmax (rows live across 16-lane group)
    float scale[8];
    #pragma unroll
    for (int i = 0; i < 8; ++i) {
      float v0 = scf[0].f[i] * taub;
      float v1 = scf[1].f[i] * taub;
      if (HASD) {
        v0 += delta[b * S + s0 + l16];
        v1 += delta[b * S + s0 + 16 + l16];
      }
      v0 *= rsE; v1 *= rsE;
      int grow = l0 + wv * 16 + i + 8 * half;
      if (CAUSAL) {
        if (s0 + l16 > grow)      v0 = -INFINITY;
        if (s0 + 16 + l16 > grow) v1 = -INFINITY;
      }
      float tm = fmaxf(v0, v1);
      for (int m = 1; m < 16; m <<= 1) tm = fmaxf(tm, __shfl_xor(tm, m, 32));
      float nm = fmaxf(rmax[i], tm);
      float sc = (nm == -INFINITY) ? 1.f : __expf(rmax[i] - nm);
      float p0 = (nm == -INFINITY) ? 0.f : __expf(v0 - nm);
      float p1 = (nm == -INFINITY) ? 0.f : __expf(v1 - nm);
      float ts = p0 + p1;
      for (int m = 1; m < 16; m <<= 1) ts += __shfl_xor(ts, m, 32);
      rsum[i] = rsum[i] * sc + ts;
      rmax[i] = nm;
      scale[i] = sc;
      int mrow = i + 8 * half;
      sP[wv][mrow * 40 + l16]      = (_Float16)p0;
      sP[wv][mrow * 40 + 16 + l16] = (_Float16)p1;
    }
    #pragma unroll
    for (int nf = 0; nf < 4; ++nf)
      #pragma unroll
      for (int i = 0; i < 8; ++i) o[nf].f[i] *= scale[i];
    __syncthreads();   // orders sP writes/reads; V buffer untouched by async path
    // O += P(16x32) @ V(32x64)
    AF pf;
    #pragma unroll
    for (int i = 0; i < 8; ++i) {
      int k = 2 * i + 8 * half + (i >= 4 ? 8 : 0);
      pf.u[i] = *(const unsigned*)(&sP[wv][0] + l16 * 40 + k);
    }
    #pragma unroll
    for (int nf = 0; nf < 4; ++nf) {
      AF vf;
      #pragma unroll
      for (int i = 0; i < 8; ++i) {
        int k = 2 * i + 8 * half + (i >= 4 ? 8 : 0);
        vf.u[i] = *(const unsigned*)(&sVt[cur][(nf * 16 + l16) * 40 + k]);
      }
      o[nf].v = WMMA_F16(pf.v, vf.v, o[nf].v);
    }
  }
  // normalize and store (f16, layout [B, Lq, H*64])
  #pragma unroll
  for (int nf = 0; nf < 4; ++nf)
    #pragma unroll
    for (int i = 0; i < 8; ++i) {
      int grow = l0 + wv * 16 + i + 8 * half;
      float rs = rsum[i];
      if (!(rs > 0.f)) rs = 1.f;
      Oo[((size_t)(b * Lq + grow)) * D + h * 64 + nf * 16 + l16] =
          (_Float16)(o[nf].f[i] / rs);
    }
}

// ------------------------- layernorm (row = 512) ---------------------------
__global__ __launch_bounds__(128)
void k_ln(const float* __restrict__ in, const float* __restrict__ g,
          const float* __restrict__ bta, float* __restrict__ outf,
          _Float16* __restrict__ outh) {
  __shared__ float red[128];
  const int row = blockIdx.x, tid = threadIdx.x;
  const float* x = in + (size_t)row * 512;
  float v[4], s = 0.f, s2 = 0.f;
  #pragma unroll
  for (int j = 0; j < 4; ++j) {
    v[j] = x[tid + j * 128];
    s += v[j]; s2 += v[j] * v[j];
  }
  red[tid] = s; __syncthreads();
  for (int off = 64; off > 0; off >>= 1) { if (tid < off) red[tid] += red[tid + off]; __syncthreads(); }
  float mean = red[0] / 512.f; __syncthreads();
  red[tid] = s2; __syncthreads();
  for (int off = 64; off > 0; off >>= 1) { if (tid < off) red[tid] += red[tid + off]; __syncthreads(); }
  float var = red[0] / 512.f - mean * mean;
  float inv = rsqrtf(var + 1e-5f);
  #pragma unroll
  for (int j = 0; j < 4; ++j) {
    int d = tid + j * 128;
    float y = (v[j] - mean) * inv * g[d] + bta[d];
    if (outf) outf[(size_t)row * 512 + d] = y;
    if (outh) outh[(size_t)row * 512 + d] = (_Float16)y;
  }
}

// ------------------------- final head --------------------------------------
__global__ __launch_bounds__(128)
void k_head(const float* __restrict__ X, const float* __restrict__ Wp,
            const float* __restrict__ bp, const float* __restrict__ xlag,
            float* __restrict__ out) {
  const int tid = threadIdx.x, lane = tid & 31, wv = tid >> 5;
  const int r = blockIdx.x * 4 + wv;     // 0 .. 16383
  const int b = r >> 9, t = r & 511;
  const float* xr = X + ((size_t)(b * 768 + 256 + t)) * 512;
  float s = 0.f;
  #pragma unroll
  for (int j = 0; j < 16; ++j) s += xr[lane + j * 32] * Wp[lane + j * 32];
  for (int m = 1; m < 32; m <<= 1) s += __shfl_xor(s, m, 32);
  s += bp[0];
  if (lane < 7)
    out[((size_t)(b * 512) + t) * 7 + lane] =
        s + xlag[((size_t)(b * 512) + t) * 7 + lane];
}

// ===========================================================================
extern "C" void kernel_launch(void* const* d_in, const int* in_sizes, int n_in,
                              void* d_out, int out_size, void* d_ws, size_t ws_size,
                              hipStream_t stream) {
  (void)in_sizes; (void)n_in; (void)out_size; (void)ws_size;
  const int ME = 32 * 512, MD = 32 * 768;

  auto F = [&](int i) -> const float* { return (const float*)d_in[i]; };
  const float* x_enc = F(0);
  const float* y_dec = F(1);

  // ---- bump allocator over workspace ----
  char* ws = (char*)d_ws;
  size_t off = 0;
  auto alloc = [&](size_t bytes) -> char* {
    char* p = ws + off;
    off += (bytes + 255) & ~(size_t)255;
    return p;
  };

  float*    Xf  = (float*)alloc((size_t)MD * 512 * 4);
  float*    Yf  = (float*)alloc((size_t)MD * 512 * 4);
  _Float16* X16 = (_Float16*)alloc((size_t)MD * 512 * 2);
  _Float16* Q16 = (_Float16*)alloc((size_t)MD * 512 * 2);
  _Float16* K16 = (_Float16*)alloc((size_t)MD * 512 * 2);
  _Float16* V16 = (_Float16*)alloc((size_t)MD * 512 * 2);
  _Float16* A16 = (_Float16*)alloc((size_t)MD * 512 * 2);
  _Float16* H16 = (_Float16*)alloc((size_t)MD * 2048 * 2);
  _Float16* E16 = (_Float16*)alloc((size_t)ME * 512 * 2);
  float* xdash    = (float*)alloc((size_t)32 * 512 * 7 * 4);
  float* xdashdec = (float*)alloc((size_t)32 * 768 * 7 * 4);
  float* xlag     = (float*)alloc((size_t)32 * 512 * 7 * 4);
  float* meanb    = (float*)alloc(32 * 7 * 4);
  float* stdb     = (float*)alloc(32 * 7 * 4);
  float* tauB     = (float*)alloc(32 * 4);
  float* deltaB   = (float*)alloc((size_t)32 * 512 * 4);

  auto cvt = [&](int idx, size_t n) -> const _Float16* {
    _Float16* dst = (_Float16*)alloc(n * sizeof(_Float16));
    int blocks = (int)((n + 255) / 256);
    if (blocks > 4096) blocks = 4096;
    k_f2h<<<blocks, 256, 0, stream>>>((const float*)d_in[idx], dst, (int)n);
    return dst;
  };

  struct AttW { const _Float16 *wq, *wk, *wv, *wo; const float *bq, *bk, *bv, *bo; };
  auto attw = [&](int base) -> AttW {
    AttW a;
    a.wq = cvt(base + 0, 512 * 512); a.wk = cvt(base + 2, 512 * 512);
    a.wv = cvt(base + 4, 512 * 512); a.wo = cvt(base + 6, 512 * 512);
    a.bq = F(base + 1); a.bk = F(base + 3); a.bv = F(base + 5); a.bo = F(base + 7);
    return a;
  };

  // weights -> f16 (deterministic, each launch)
  AttW encA[2] = { attw(6), attw(22) };
  const _Float16* encF1[2] = { cvt(14, 512 * 2048), cvt(30, 512 * 2048) };
  const _Float16* encF2[2] = { cvt(16, 2048 * 512), cvt(32, 2048 * 512) };
  AttW saw = attw(40), caw = attw(48);
  const _Float16* decF1 = cvt(56, 512 * 2048);
  const _Float16* decF2 = cvt(58, 2048 * 512);

  // ---- launch helpers (compile-time epilogue selection) ----
  auto gemm_h16 = [&](const _Float16* Aa, const _Float16* Ww, const float* bias,
                      _Float16* outh, int M, int N, int K, int relu) {
    dim3 g(N / 64, M / 128);
    if (relu)
      k_gemm<0, 1, 0, 1><<<g, 128, 0, stream>>>(Aa, Ww, bias, nullptr, nullptr, outh, M, N, K);
    else
      k_gemm<0, 0, 0, 1><<<g, 128, 0, stream>>>(Aa, Ww, bias, nullptr, nullptr, outh, M, N, K);
  };
  auto gemm_res = [&](const _Float16* Aa, const _Float16* Ww, const float* bias,
                      const float* resid, float* outf, int M, int N, int K) {
    dim3 g(N / 64, M / 128);
    k_gemm<1, 0, 1, 0><<<g, 128, 0, stream>>>(Aa, Ww, bias, resid, outf, nullptr, M, N, K);
  };
  auto ln = [&](const float* in, const float* g_, const float* b_,
                float* outf, _Float16* outh, int M) {
    k_ln<<<M, 128, 0, stream>>>(in, g_, b_, outf, outh);
  };
  auto attn = [&](int Lq, int S, int causal, const float* dlt) {
    dim3 g(Lq / 64, 32 * 8);
    if (causal)
      k_attn<1, 0><<<g, 128, 0, stream>>>(Q16, K16, V16, A16, tauB, nullptr, 8, Lq, S);
    else
      k_attn<0, 1><<<g, 128, 0, stream>>>(Q16, K16, V16, A16, tauB, dlt, 8, Lq, S);
  };
  auto attn_block = [&](const AttW& w, const _Float16* xq, const _Float16* xkv,
                        int Mq, int Mkv, int Lq, int S, int causal,
                        const float* dlt, const float* lng, const float* lnb) {
    gemm_h16(xq,  w.wq, w.bq, Q16, Mq,  512, 512, 0);
    gemm_h16(xkv, w.wk, w.bk, K16, Mkv, 512, 512, 0);
    gemm_h16(xkv, w.wv, w.bv, V16, Mkv, 512, 512, 0);
    attn(Lq, S, causal, dlt);
    gemm_res(A16, w.wo, w.bo, Xf, Yf, Mq, 512, 512);
    ln(Yf, lng, lnb, Xf, X16, Mq);
  };
  auto ffn_block = [&](const _Float16* wf1, const float* bf1, const _Float16* wf2,
                       const float* bf2, const float* lng, const float* lnb, int M) {
    gemm_h16(X16, wf1, bf1, H16, M, 2048, 512, 1);
    gemm_res(H16, wf2, bf2, Xf, Yf, M, 512, 2048);
    ln(Yf, lng, lnb, Xf, X16, M);
  };

  // ---- preprocessing + projectors ----
  k_prep<<<1, 256, 0, stream>>>(x_enc, y_dec, xdash, xdashdec, xlag, meanb, stdb);
  k_proj<<<32, 256, 0, stream>>>(x_enc, stdb, F(70), F(71), F(72), F(73), F(74), F(75),
                                 tauB, 1, 1);                    // tau = exp(proj)
  k_proj<<<32, 256, 0, stream>>>(x_enc, meanb, F(76), F(77), F(78), F(79), F(80), F(81),
                                 deltaB, 512, 0);                // delta [B,512]

  // ---- encoder ----
  k_embed<<<2048, 256, 0, stream>>>(xdash, F(2), F(3), Xf, X16, ME, 512);
  for (int li = 0; li < 2; ++li) {
    int base = 6 + 16 * li;
    attn_block(encA[li], X16, X16, ME, ME, 512, 512, 0, deltaB,
               F(base + 12), F(base + 13));
    ffn_block(encF1[li], F(base + 9), encF2[li], F(base + 11),
              F(base + 14), F(base + 15), ME);
  }
  ln(Xf, F(38), F(39), Yf, E16, ME);   // enc final norm -> E16 (K/V source for cross)

  // ---- decoder ----
  k_embed<<<2048, 256, 0, stream>>>(xdashdec, F(4), F(5), Xf, X16, MD, 768);
  attn_block(saw, X16, X16, MD, MD, 768, 768, 1, nullptr, F(60), F(61));   // causal SA
  attn_block(caw, X16, E16, MD, ME, 768, 512, 0, deltaB, F(62), F(63));    // cross A
  ffn_block(decF1, F(57), decF2, F(59), F(64), F(65), MD);
  ln(Xf, F(66), F(67), Yf, nullptr, MD);                                   // dec_norm

  // ---- head: projection + lagged add ----
  k_head<<<4096, 128, 0, stream>>>(Yf, F(68), F(69), xlag, (float*)d_out);
}
